// MiniMaxText01LightningAttention_13907104105120
// MI455X (gfx1250) — compile-verified, matrix operand order checked by reading, not run on hardware
//
#include <hip/hip_runtime.h>
#include <stdint.h>
#include <stddef.h>

#define B_    2
#define S_    4096
#define H_    2048
#define NH_   16
#define HD_   128
#define BLK_  256
#define NCHUNK (S_ / BLK_)
#define MTOT  (B_ * S_)      // 8192 rows
#define KDIM  H_             // 2048 reduction dim for all GEMMs
#define EPS_  1e-5f

typedef __bf16 bf16;
typedef __attribute__((ext_vector_type(16))) __bf16      v16bf;
typedef __attribute__((ext_vector_type(8)))  float       v8f;
typedef __attribute__((ext_vector_type(4)))  unsigned int u32x4;

union Frag { struct { u32x4 lo, hi; } q; __bf16 e[16]; v16bf v; };
union Acc  { v8f v; float e[8]; };

// Load a 16-element bf16 fragment as two contiguous 8-element (16B) runs.
// Matches CDNA5 16-bit A/B layout: lane L%16 = row, runs at k0+hi8 and k0+16+hi8.
static __device__ __forceinline__ v16bf ldfrag(const bf16* r1, const bf16* r2) {
    Frag f;
    f.q.lo = *reinterpret_cast<const u32x4*>(r1);
    f.q.hi = *reinterpret_cast<const u32x4*>(r2);
    return f.v;
}

static __device__ __forceinline__ v8f wmma_bf16(v16bf a, v16bf b, v8f c) {
    return __builtin_amdgcn_wmma_f32_16x16x32_bf16(false, a, false, b, (short)0, c,
                                                   false, false);
}

// Async Global -> LDS copy (CDNA5 GLOBAL_LOAD_ASYNC_TO_LDS_B128, ASYNCcnt-tracked).
// VDST VGPR carries per-lane LDS byte address (low 32 bits of generic shared ptr).
static __device__ __forceinline__ void async_copy_b128(void* lds_ptr, const void* gptr) {
    unsigned lds_off = (unsigned)(uintptr_t)lds_ptr;
    asm volatile("global_load_async_to_lds_b128 %0, %1, off"
                 :: "v"(lds_off), "v"(gptr)
                 : "memory");
}
static __device__ __forceinline__ void wait_async_le4() {
    asm volatile("s_wait_asynccnt 4" ::: "memory");
}
static __device__ __forceinline__ void wait_async_0() {
    asm volatile("s_wait_asynccnt 0" ::: "memory");
}

// ---------------------------------------------------------------- conversions
__global__ void k_cvt_bf16(const float* __restrict__ src, bf16* __restrict__ dst, int n) {
    int i = blockIdx.x * blockDim.x + threadIdx.x;
    int stride = gridDim.x * blockDim.x;
    for (; i < n; i += stride) dst[i] = (bf16)src[i];
}

// src[K][N] fp32 -> dst[N][K] bf16 (coalesced reads)
__global__ void k_transpose_bf16(const float* __restrict__ src, bf16* __restrict__ dst,
                                 int K, int N) {
    long long i = (long long)blockIdx.x * blockDim.x + threadIdx.x;
    long long tot = (long long)K * N;
    long long stride = (long long)gridDim.x * blockDim.x;
    for (; i < tot; i += stride) {
        int n = (int)(i % N), k = (int)(i / N);
        dst[(long long)n * K + k] = (bf16)src[i];
    }
}

// ---------------------------------------------------------------- unified GEMM
// C[m][n] = epilogue( A[m][:] . Bt[n][:] )  with A:[MTOT][KDIM], Bt:[N][KDIM] bf16
// Double-buffered: stage k+1 streams into LDS via async engine while stage k computes.
#define GPAD 40   // 32 + 8 pad -> 80B row stride (multiple of 16B)
#define NKSTEP (KDIM / 32)

__global__ __launch_bounds__(256) void k_gemm(
    const bf16* __restrict__ A, const bf16* __restrict__ Bt, int Ncols, int mode,
    bf16* __restrict__ qb, bf16* __restrict__ kb, bf16* __restrict__ vb,
    const float* __restrict__ mask, const float* __restrict__ onorm,
    bf16* __restrict__ gated, float* __restrict__ outp)
{
    __shared__ bf16 As[2][128 * GPAD];
    __shared__ bf16 Bs[2][128 * GPAD];

    const int t    = threadIdx.x;
    const int lane = t & 31;
    const int wave = t >> 5;
    const int wm   = wave & 3;   // 4 waves along M (32 rows each)
    const int wn   = wave >> 2;  // 2 waves along N (64 cols each)
    const int m0   = blockIdx.y * 128;
    const int n0   = blockIdx.x * 128;

    Acc c[2][4];
#pragma unroll
    for (int i = 0; i < 2; i++)
#pragma unroll
        for (int j = 0; j < 4; j++)
#pragma unroll
            for (int r = 0; r < 8; r++) c[i][j].e[r] = 0.f;

    const int lrow = t >> 1;         // 0..127
    const int lseg = (t & 1) * 16;   // 0 or 16
    const int hi8  = (lane >> 4) * 8;
    const int lm   = lane & 15;

    const bf16* gA = A  + (long long)(m0 + lrow) * KDIM + lseg;
    const bf16* gB = Bt + (long long)(n0 + lrow) * KDIM + lseg;

    // issue one 32-wide K stage (4 x b128 per thread, ASYNCcnt += 4)
    auto issue = [&](int ks, int buf) {
        const bf16* ga = gA + ks * 32;
        const bf16* gb = gB + ks * 32;
        bf16* la = &As[buf][lrow * GPAD + lseg];
        bf16* lb = &Bs[buf][lrow * GPAD + lseg];
        async_copy_b128(la,     ga);
        async_copy_b128(la + 8, ga + 8);
        async_copy_b128(lb,     gb);
        async_copy_b128(lb + 8, gb + 8);
    };

    issue(0, 0);
    issue(1, 1);

    for (int ks = 0; ks < NKSTEP; ++ks) {
        const int buf = ks & 1;
        if (ks + 1 < NKSTEP) wait_async_le4();   // stage ks landed, ks+1 in flight
        else                 wait_async_0();
        __syncthreads();                          // all waves' data visible

        v16bf afr[2], bfr[4];
#pragma unroll
        for (int i = 0; i < 2; i++) {
            const bf16* base = &As[buf][(wm * 32 + i * 16 + lm) * GPAD];
            afr[i] = ldfrag(base + hi8, base + 16 + hi8);
        }
#pragma unroll
        for (int j = 0; j < 4; j++) {
            const bf16* base = &Bs[buf][(wn * 64 + j * 16 + lm) * GPAD];
            bfr[j] = ldfrag(base + hi8, base + 16 + hi8);
        }
#pragma unroll
        for (int i = 0; i < 2; i++)
#pragma unroll
            for (int j = 0; j < 4; j++)
                c[i][j].v = wmma_bf16(afr[i], bfr[j], c[i][j].v);

        __syncthreads();                          // everyone done reading buf
        if (ks + 2 < NKSTEP) issue(ks + 2, buf);  // refill freed buffer
    }

    // epilogue: C element (vgpr r, lane L) -> row = hi*8 + r, col = L%16
    const int rb = hi8;
#pragma unroll
    for (int i = 0; i < 2; i++) {
#pragma unroll
        for (int j = 0; j < 4; j++) {
#pragma unroll
            for (int r = 0; r < 8; r++) {
                int row = m0 + wm * 32 + i * 16 + rb + r;
                int col = n0 + wn * 64 + j * 16 + lm;
                float x = c[i][j].e[r];
                if (mode == 0) {                       // qkv: silu + scatter heads
                    float sv = x / (1.f + __expf(-x));
                    int which = col / (NH_ * HD_);
                    int rem   = col % (NH_ * HD_);
                    int h = rem / HD_, d = rem % HD_;
                    int bb = row / S_, ss = row % S_;
                    long long dst = (((long long)bb * NH_ + h) * S_ + ss) * HD_ + d;
                    if (which == 0)      qb[dst] = (bf16)sv;
                    else if (which == 1) kb[dst] = (bf16)sv;
                    else                 vb[dst] = (bf16)(sv * mask[bb * S_ + ss]);
                } else if (mode == 1) {                // gate: sigmoid * o_norm
                    float g = 1.f / (1.f + __expf(-x));
                    long long idx = (long long)row * H_ + col;
                    gated[idx] = (bf16)(g * onorm[idx]);
                } else {                               // final output
                    outp[(long long)row * H_ + col] = x;
                }
            }
        }
    }
}

// ---------------------------------------------------------------- rmsnorm (in place)
__global__ __launch_bounds__(256) void k_rmsnorm(float* __restrict__ o,
                                                 const float* __restrict__ w) {
    __shared__ float red[256];
    float* p = o + (long long)blockIdx.x * H_;
    float s = 0.f;
    for (int i = threadIdx.x; i < H_; i += 256) { float x = p[i]; s += x * x; }
    red[threadIdx.x] = s;
    __syncthreads();
    for (int k = 128; k > 0; k >>= 1) {
        if (threadIdx.x < k) red[threadIdx.x] += red[threadIdx.x + k];
        __syncthreads();
    }
    float scale = rsqrtf(red[0] / (float)H_ + EPS_);
    for (int i = threadIdx.x; i < H_; i += 256) p[i] = p[i] * scale * w[i];
}

// ---------------------------------------------------------------- lightning attention
#define NP  264   // 256 + 8 pad (row stride 528B, 16B multiple)
#define KVP 136   // 128 + 8 pad (row stride 272B, 16B multiple)

__global__ __launch_bounds__(256) void k_attn(
    const bf16* __restrict__ q, const bf16* __restrict__ k,
    const bf16* __restrict__ v, const float* __restrict__ slope,
    float* __restrict__ obuf)
{
    extern __shared__ char smem[];
    bf16*  kiT = (bf16*)smem;                                          // [128][NP]  (k*kdec)^T
    bf16*  viT = (bf16*)(smem + 128 * NP * 2);                         // [128][NP]  v^T
    float* kv  = (float*)(smem + 2 * 128 * NP * 2);                    // [128][128] fp32 state
    bf16*  kvT = (bf16*)(smem + 2 * 128 * NP * 2 + 128 * 128 * 4);     // [128][KVP] kv^T bf16
    bf16*  atn = (bf16*)(smem + 2 * 128 * NP * 2 + 128 * 128 * 4 + 128 * KVP * 2); // [64][NP]

    const int bh = blockIdx.x;
    const int b = bh / NH_, h = bh % NH_;
    const float s    = slope[h];
    const float bdec = __expf(-s * (float)BLK_);

    const int t = threadIdx.x, lane = t & 31, wave = t >> 5;
    const int hi8 = (lane >> 4) * 8;
    const int lm  = lane & 15;

    const bf16* qg = q + (long long)bh * S_ * HD_;
    const bf16* kg = k + (long long)bh * S_ * HD_;
    const bf16* vg = v + (long long)bh * S_ * HD_;

    for (int i = t; i < 128 * 128; i += 256) kv[i] = 0.f;
    for (int i = t; i < 128 * KVP; i += 256) kvT[i] = (bf16)0.f;
    __syncthreads();

    for (int c = 0; c < NCHUNK; ++c) {
        const int sb = c * BLK_;
        // stage (k * k_decay)^T and v^T
        for (int i = t; i < BLK_ * HD_; i += 256) {
            int n = i >> 7, d = i & 127;
            float kd = __expf(-s * (float)(BLK_ - 1 - n));
            kiT[d * NP + n] = (bf16)((float)kg[(long long)(sb + n) * HD_ + d] * kd);
            viT[d * NP + n] = vg[(long long)(sb + n) * HD_ + d];
        }
        __syncthreads();

        for (int sub = 0; sub < 4; ++sub) {
            const int msub = sub * 64;
            // ---- attn = q @ k^T, diag decay -> atn (bf16) ----
            {
                const int awm = wave >> 1;   // 4 waves x 16 rows
                const int awn = wave & 1;    // 2 waves x 128 cols
                Acc ac[8];
#pragma unroll
                for (int j = 0; j < 8; j++)
#pragma unroll
                    for (int r = 0; r < 8; r++) ac[j].e[r] = 0.f;

                const int mrow = msub + awm * 16 + lm;
                const bf16* arow = qg + (long long)(sb + mrow) * HD_;
#pragma unroll
                for (int k0 = 0; k0 < HD_; k0 += 32) {
                    v16bf af = ldfrag(arow + k0 + hi8, arow + k0 + 16 + hi8);
#pragma unroll
                    for (int j = 0; j < 8; j++) {
                        int nn = awn * 128 + j * 16 + lm;
                        const bf16* brow = kg + (long long)(sb + nn) * HD_;
                        v16bf bfr = ldfrag(brow + k0 + hi8, brow + k0 + 16 + hi8);
                        ac[j].v = wmma_bf16(af, bfr, ac[j].v);
                    }
                }
#pragma unroll
                for (int j = 0; j < 8; j++) {
                    int nn = awn * 128 + j * 16 + lm;
#pragma unroll
                    for (int r = 0; r < 8; r++) {
                        int mm = msub + awm * 16 + hi8 + r;
                        float dd = (mm >= nn) ? __expf(-s * (float)(mm - nn)) : 0.f;
                        atn[(mm - msub) * NP + nn] = (bf16)(ac[j].e[r] * dd);
                    }
                }
            }
            __syncthreads();
            // ---- o = (q * q_decay) @ kv^T + atn @ v^T ----
            {
                const int owm = wave >> 1;   // 4 waves x 16 rows
                const int own = wave & 1;    // 2 waves x 64 e-cols
                Acc oc[4];
#pragma unroll
                for (int j = 0; j < 4; j++)
#pragma unroll
                    for (int r = 0; r < 8; r++) oc[j].e[r] = 0.f;

                const int mrow = msub + owm * 16 + lm;
                const float qd = __expf(-s * (float)(mrow + 1));
                const bf16* arow = qg + (long long)(sb + mrow) * HD_;
#pragma unroll
                for (int k0 = 0; k0 < HD_; k0 += 32) {
                    Frag f;
                    f.q.lo = *reinterpret_cast<const u32x4*>(arow + k0 + hi8);
                    f.q.hi = *reinterpret_cast<const u32x4*>(arow + k0 + 16 + hi8);
#pragma unroll
                    for (int x = 0; x < 16; x++) f.e[x] = (bf16)((float)f.e[x] * qd);
#pragma unroll
                    for (int j = 0; j < 4; j++) {
                        int ee = own * 64 + j * 16 + lm;
                        const bf16* brow = &kvT[ee * KVP];
                        v16bf bfr = ldfrag(brow + k0 + hi8, brow + k0 + 16 + hi8);
                        oc[j].v = wmma_bf16(f.v, bfr, oc[j].v);
                    }
                }
                const bf16* arow2 = &atn[(owm * 16 + lm) * NP];
#pragma unroll
                for (int k0 = 0; k0 < BLK_; k0 += 32) {
                    v16bf af = ldfrag(arow2 + k0 + hi8, arow2 + k0 + 16 + hi8);
#pragma unroll
                    for (int j = 0; j < 4; j++) {
                        int ee = own * 64 + j * 16 + lm;
                        const bf16* brow = &viT[ee * NP];
                        v16bf bfr = ldfrag(brow + k0 + hi8, brow + k0 + 16 + hi8);
                        oc[j].v = wmma_bf16(af, bfr, oc[j].v);
                    }
                }
#pragma unroll
                for (int j = 0; j < 4; j++) {
                    int ee = own * 64 + j * 16 + lm;
#pragma unroll
                    for (int r = 0; r < 8; r++) {
                        int mm = msub + owm * 16 + hi8 + r;
                        long long ss = sb + mm;
                        obuf[((long long)b * S_ + ss) * H_ + h * HD_ + ee] = oc[j].e[r];
                    }
                }
            }
            __syncthreads();
        }
        // ---- kv = bdec*kv + (k*kdec)^T @ v ----
        {
            const int wd = wave >> 1;   // 4 waves x 32 d-rows
            const int we = wave & 1;    // 2 waves x 64 e-cols
            Acc cv[2][4];
#pragma unroll
            for (int i = 0; i < 2; i++)
#pragma unroll
                for (int j = 0; j < 4; j++)
#pragma unroll
                    for (int r = 0; r < 8; r++) {
                        int dd = wd * 32 + i * 16 + hi8 + r;
                        int ee = we * 64 + j * 16 + lm;
                        cv[i][j].e[r] = kv[dd * 128 + ee] * bdec;
                    }
#pragma unroll
            for (int k0 = 0; k0 < BLK_; k0 += 32) {
                v16bf af[2];
#pragma unroll
                for (int i = 0; i < 2; i++) {
                    const bf16* arow = &kiT[(wd * 32 + i * 16 + lm) * NP];
                    af[i] = ldfrag(arow + k0 + hi8, arow + k0 + 16 + hi8);
                }
#pragma unroll
                for (int j = 0; j < 4; j++) {
                    int ee = we * 64 + j * 16 + lm;
                    const bf16* brow = &viT[ee * NP];
                    v16bf bfr = ldfrag(brow + k0 + hi8, brow + k0 + 16 + hi8);
#pragma unroll
                    for (int i = 0; i < 2; i++) cv[i][j].v = wmma_bf16(af[i], bfr, cv[i][j].v);
                }
            }
#pragma unroll
            for (int i = 0; i < 2; i++)
#pragma unroll
                for (int j = 0; j < 4; j++)
#pragma unroll
                    for (int r = 0; r < 8; r++) {
                        int dd = wd * 32 + i * 16 + hi8 + r;
                        int ee = we * 64 + j * 16 + lm;
                        kv[dd * 128 + ee] = cv[i][j].e[r];
                    }
        }
        __syncthreads();
        for (int i = t; i < 128 * 128; i += 256) {
            int ee = i >> 7, dd = i & 127;
            kvT[ee * KVP + dd] = (bf16)kv[dd * 128 + ee];
        }
        __syncthreads();
    }
}

// ---------------------------------------------------------------- launch
extern "C" void kernel_launch(void* const* d_in, const int* in_sizes, int n_in,
                              void* d_out, int out_size, void* d_ws, size_t ws_size,
                              hipStream_t stream) {
    const float* hidden = (const float*)d_in[0];
    const float* mask   = (const float*)d_in[1];
    const float* slope  = (const float*)d_in[2];
    const float* w_qkv  = (const float*)d_in[3];
    const float* w_gate = (const float*)d_in[4];
    const float* w_out  = (const float*)d_in[5];
    const float* normw  = (const float*)d_in[6];
    float* outp = (float*)d_out;

    char* ws = (char*)d_ws;
    size_t off = 0;
    auto alloc = [&](size_t bytes) -> char* {
        char* p = ws + off;
        off += bytes;
        off = (off + 255) & ~(size_t)255;
        return p;
    };
    bf16* hid_b = (bf16*)alloc((size_t)MTOT * KDIM * 2);
    bf16* wqkvT = (bf16*)alloc((size_t)3 * NH_ * HD_ * KDIM * 2);
    bf16* wgT   = (bf16*)alloc((size_t)H_ * KDIM * 2);
    bf16* woT   = (bf16*)alloc((size_t)H_ * KDIM * 2);
    bf16* qb    = (bf16*)alloc((size_t)B_ * NH_ * S_ * HD_ * 2);
    bf16* kb    = (bf16*)alloc((size_t)B_ * NH_ * S_ * HD_ * 2);
    bf16* vb    = (bf16*)alloc((size_t)B_ * NH_ * S_ * HD_ * 2);
    float* ob   = (float*)alloc((size_t)MTOT * H_ * 4);
    bf16* gated = (bf16*)alloc((size_t)MTOT * H_ * 2);

    k_cvt_bf16<<<1024, 256, 0, stream>>>(hidden, hid_b, MTOT * KDIM);
    k_transpose_bf16<<<1024, 256, 0, stream>>>(w_qkv, wqkvT, KDIM, 3 * NH_ * HD_);
    k_transpose_bf16<<<1024, 256, 0, stream>>>(w_gate, wgT, KDIM, H_);
    k_transpose_bf16<<<1024, 256, 0, stream>>>(w_out, woT, H_, H_);

    dim3 g0(3 * NH_ * HD_ / 128, MTOT / 128);
    k_gemm<<<g0, 256, 0, stream>>>(hid_b, wqkvT, 3 * NH_ * HD_, 0,
                                   qb, kb, vb, mask, nullptr, nullptr, nullptr);

    size_t attn_lds = (size_t)(2 * 128 * NP * 2 + 128 * 128 * 4 + 128 * KVP * 2 + 64 * NP * 2);
    (void)hipFuncSetAttribute((const void*)k_attn,
                              hipFuncAttributeMaxDynamicSharedMemorySize, (int)attn_lds);
    k_attn<<<B_ * NH_, 256, attn_lds, stream>>>(qb, kb, vb, slope, ob);

    k_rmsnorm<<<MTOT, 256, 0, stream>>>(ob, normw);

    dim3 g1(H_ / 128, MTOT / 128);
    k_gemm<<<g1, 256, 0, stream>>>(hid_b, wgT, H_, 1,
                                   nullptr, nullptr, nullptr, nullptr, ob, gated, nullptr);
    k_gemm<<<g1, 256, 0, stream>>>(gated, woT, H_, 2,
                                   nullptr, nullptr, nullptr, nullptr, nullptr, nullptr, outp);
}
